// Memory_45440753992368
// MI455X (gfx1250) — compile-verified, matrix operand order checked by reading, not run on hardware
//
#include <hip/hip_runtime.h>

typedef __bf16 bf16_t;
typedef bf16_t v16bf __attribute__((ext_vector_type(16)));
typedef float  v8f   __attribute__((ext_vector_type(8)));

union ABFrag { v16bf v; uint4 u[2]; };

#define BS     32768
#define MSLOTS 2048
#define DDIM   256

__device__ __forceinline__ float wave_sum(float v) {
#pragma unroll
    for (int o = 16; o > 0; o >>= 1) v += __shfl_xor(v, o, 32);
    return v;
}

// ---------------- init ----------------
__global__ void k_init(float* qupd, float* loss) {
    int idx = blockIdx.x * blockDim.x + threadIdx.x;
    if (idx < MSLOTS * DDIM) qupd[idx] = 0.f;
    if (idx < 2) loss[idx] = 0.f;
}

// ------------- normalize query: q fp32, q bf16, left half of update_query -------------
__global__ void __launch_bounds__(256) k_norm_query(const float* __restrict__ query,
        float* __restrict__ qf32, bf16_t* __restrict__ qbf, float* __restrict__ uq) {
    int lane = threadIdx.x & 31, wave = threadIdx.x >> 5;
    int j = blockIdx.x * 8 + wave;
    const float* row = query + (size_t)j * DDIM;
    int c = lane * 8;
    float4 a = *(const float4*)(row + c);
    float4 b = *(const float4*)(row + c + 4);
    float s = a.x*a.x + a.y*a.y + a.z*a.z + a.w*a.w
            + b.x*b.x + b.y*b.y + b.z*b.z + b.w*b.w;
    s = wave_sum(s);
    float inv = 1.f / fmaxf(sqrtf(s), 1e-12f);
    float v[8] = {a.x*inv, a.y*inv, a.z*inv, a.w*inv, b.x*inv, b.y*inv, b.z*inv, b.w*inv};
    float4* qo = (float4*)(qf32 + (size_t)j * DDIM + c);
    qo[0] = make_float4(v[0], v[1], v[2], v[3]);
    qo[1] = make_float4(v[4], v[5], v[6], v[7]);
    float4* uo = (float4*)(uq + (size_t)j * (2 * DDIM) + c);
    uo[0] = make_float4(v[0], v[1], v[2], v[3]);
    uo[1] = make_float4(v[4], v[5], v[6], v[7]);
    bf16_t* qb = qbf + (size_t)j * DDIM + c;
#pragma unroll
    for (int i = 0; i < 8; ++i) qb[i] = (bf16_t)v[i];
}

// ------------- memory -> bf16 row-major + bf16 transposed -------------
__global__ void __launch_bounds__(256) k_prep_mem(const float* __restrict__ memory,
        bf16_t* __restrict__ membf, bf16_t* __restrict__ memT) {
    int idx = blockIdx.x * 256 + threadIdx.x;       // over M*D
    int r = idx >> 8, c = idx & 255;
    float v = memory[idx];
    membf[idx] = (bf16_t)v;
    memT[(size_t)c * MSLOTS + r] = (bf16_t)v;
}

// ------------- GEMM1: score = q_bf16 @ mem_bf16^T  (WMMA bf16, fp32 acc) -------------
__global__ void __launch_bounds__(256) k_gemm_score(const bf16_t* __restrict__ qbf,
        const bf16_t* __restrict__ membf, float* __restrict__ score) {
    int lane = threadIdx.x & 31, wave = threadIdx.x >> 5;
    int m0 = (blockIdx.y * 8 + wave) * 16;
    int n0 = blockIdx.x * 64;
    int kb = (lane >> 4) * 8;                       // lanes 0-15: K 0..7(+16..23); 16-31: 8..15(+24..31)
    const bf16_t* ap = qbf   + (size_t)(m0 + (lane & 15)) * DDIM + kb;
    const bf16_t* bp = membf + (size_t)(n0 + (lane & 15)) * DDIM + kb;
    v8f acc0 = {}, acc1 = {}, acc2 = {}, acc3 = {};
#pragma unroll
    for (int ks = 0; ks < DDIM / 32; ++ks) {
        int k0 = ks * 32;
        ABFrag A, B0, B1, B2, B3;
        A.u[0]  = *(const uint4*)(ap + k0);
        A.u[1]  = *(const uint4*)(ap + k0 + 16);
        B0.u[0] = *(const uint4*)(bp + k0);
        B0.u[1] = *(const uint4*)(bp + k0 + 16);
        B1.u[0] = *(const uint4*)(bp + 16 * DDIM + k0);
        B1.u[1] = *(const uint4*)(bp + 16 * DDIM + k0 + 16);
        B2.u[0] = *(const uint4*)(bp + 32 * DDIM + k0);
        B2.u[1] = *(const uint4*)(bp + 32 * DDIM + k0 + 16);
        B3.u[0] = *(const uint4*)(bp + 48 * DDIM + k0);
        B3.u[1] = *(const uint4*)(bp + 48 * DDIM + k0 + 16);
        acc0 = __builtin_amdgcn_wmma_f32_16x16x32_bf16(false, A.v, false, B0.v, (short)0, acc0, false, false);
        acc1 = __builtin_amdgcn_wmma_f32_16x16x32_bf16(false, A.v, false, B1.v, (short)0, acc1, false, false);
        acc2 = __builtin_amdgcn_wmma_f32_16x16x32_bf16(false, A.v, false, B2.v, (short)0, acc2, false, false);
        acc3 = __builtin_amdgcn_wmma_f32_16x16x32_bf16(false, A.v, false, B3.v, (short)0, acc3, false, false);
    }
    int ro = m0 + ((lane >> 4) << 3);
    int co = n0 + (lane & 15);
#pragma unroll
    for (int r = 0; r < 8; ++r) {
        size_t rb = (size_t)(ro + r) * MSLOTS + co;
        score[rb]      = acc0[r];
        score[rb + 16] = acc1[r];
        score[rb + 32] = acc2[r];
        score[rb + 48] = acc3[r];
    }
}

// ------------- column softmax stats: partials over 16 row-chunks -------------
__global__ void __launch_bounds__(256) k_col_partial(const float* __restrict__ score,
        float* __restrict__ pmx, float* __restrict__ ps) {
    int c = blockIdx.x * 256 + threadIdx.x;
    const float* p = score + (size_t)(blockIdx.y * (BS / 16)) * MSLOTS + c;
    float mx = -3.4e38f, s = 0.f;
    for (int r = 0; r < BS / 16; ++r) {
        float v = p[(size_t)r * MSLOTS];
        float nm = fmaxf(mx, v);
        s = s * __expf(mx - nm) + __expf(v - nm);
        mx = nm;
    }
    pmx[blockIdx.y * MSLOTS + c] = mx;
    ps [blockIdx.y * MSLOTS + c] = s;
}

__global__ void k_col_merge(const float* __restrict__ pmx, const float* __restrict__ ps,
        float* __restrict__ cms, float* __restrict__ csum) {
    int c = blockIdx.x * 256 + threadIdx.x;
    float mx = -3.4e38f, s = 0.f;
    for (int b = 0; b < 16; ++b) {
        float om = pmx[b * MSLOTS + c], os = ps[b * MSLOTS + c];
        float nm = fmaxf(mx, om);
        s = s * __expf(mx - nm) + os * __expf(om - nm);
        mx = nm;
    }
    cms[c] = mx; csum[c] = s;
}

// ------------- row stats: online softmax + top2 (wave per row) -------------
__global__ void __launch_bounds__(256) k_row_stats(const float* __restrict__ score,
        float* __restrict__ rmax, float* __restrict__ rsum,
        int* __restrict__ t1, int* __restrict__ t2) {
    int lane = threadIdx.x & 31, wave = threadIdx.x >> 5;
    int j = blockIdx.x * 8 + wave;
    const float* row = score + (size_t)j * MSLOTS;
    float mx = -3.4e38f, s = 0.f;
    float v1 = -3.4e38f, v2 = -3.4e38f; int i1 = 0, i2 = 0;
    for (int i = 0; i < MSLOTS / 32; ++i) {
        int c = lane + i * 32;
        float v = row[c];
        float nm = fmaxf(mx, v);
        s = s * __expf(mx - nm) + __expf(v - nm);
        mx = nm;
        if (v > v1)      { v2 = v1; i2 = i1; v1 = v; i1 = c; }
        else if (v > v2) { v2 = v;  i2 = c; }
    }
#pragma unroll
    for (int o = 16; o > 0; o >>= 1) {
        float omx = __shfl_xor(mx, o, 32);
        float os  = __shfl_xor(s,  o, 32);
        float w1  = __shfl_xor(v1, o, 32);
        int   j1  = __shfl_xor(i1, o, 32);
        float w2  = __shfl_xor(v2, o, 32);
        int   j2  = __shfl_xor(i2, o, 32);
        float nm = fmaxf(mx, omx);
        s = s * __expf(mx - nm) + os * __expf(omx - nm);
        mx = nm;
        if (w1 > v1) {
            if (v1 >= w2) { v2 = v1; i2 = i1; } else { v2 = w2; i2 = j2; }
            v1 = w1; i1 = j1;
        } else if (w1 > v2) { v2 = w1; i2 = j1; }
    }
    if (lane == 0) { rmax[j] = mx; rsum[j] = s; t1[j] = i1; t2[j] = i2; }
}

// ------------- softmax write (sfxq in place + sfxm) + row argmax of sfxq -------------
__global__ void __launch_bounds__(256) k_softmax(float* __restrict__ sfxq, float* __restrict__ sfxm,
        const float* __restrict__ cms, const float* __restrict__ csum,
        const float* __restrict__ rmax, const float* __restrict__ rsum,
        float* __restrict__ wbuf, int* __restrict__ midx) {
    int lane = threadIdx.x & 31, wave = threadIdx.x >> 5;
    int j = blockIdx.x * 8 + wave;
    float rm = rmax[j], rinv = 1.f / rsum[j];
    float best = -3.4e38f; int bidx = 0;
    size_t base = (size_t)j * MSLOTS;
    for (int i = 0; i < MSLOTS / 32; ++i) {
        int c = lane + i * 32;
        float sv = sfxq[base + c];
        float qv = __expf(sv - cms[c]) / csum[c];
        float mv = __expf(sv - rm) * rinv;
        sfxq[base + c] = qv;
        sfxm[base + c] = mv;
        if (qv > best) { best = qv; bidx = c; }
    }
#pragma unroll
    for (int o = 16; o > 0; o >>= 1) {
        float ob = __shfl_xor(best, o, 32);
        int   oi = __shfl_xor(bidx, o, 32);
        if (ob > best || (ob == best && oi < bidx)) { best = ob; bidx = oi; }
    }
    if (lane == 0) { wbuf[j] = best * csum[bidx]; midx[j] = bidx; }  // colmax(sfxq)=1/csum
}

// ------------- segment sum via atomics -------------
__global__ void __launch_bounds__(256) k_segsum(const float* __restrict__ qf32,
        const float* __restrict__ wbuf, const int* __restrict__ midx, float* __restrict__ qupd) {
    int j = blockIdx.x, t = threadIdx.x;
    int mi = midx[j];
    atomicAdd(&qupd[(size_t)mi * DDIM + t], wbuf[j] * qf32[(size_t)j * DDIM + t]);
}

// ------------- updated_memory = normalize(qupd + memory) -------------
__global__ void __launch_bounds__(256) k_update_mem(const float* __restrict__ qupd,
        const float* __restrict__ memory, float* __restrict__ umem) {
    int lane = threadIdx.x & 31, wave = threadIdx.x >> 5;
    int r = blockIdx.x * 8 + wave;
    size_t base = (size_t)r * DDIM + lane * 8;
    float v[8]; float s = 0.f;
#pragma unroll
    for (int i = 0; i < 8; ++i) { v[i] = qupd[base + i] + memory[base + i]; s += v[i] * v[i]; }
    s = wave_sum(s);
    float inv = 1.f / fmaxf(sqrtf(s), 1e-12f);
#pragma unroll
    for (int i = 0; i < 8; ++i) umem[base + i] = v[i] * inv;
}

// ------------- GEMM2: concat = sfxm @ memory (A fp32->bf16 in-register, WMMA) -------------
__global__ void __launch_bounds__(256) k_gemm_read(const float* __restrict__ sfxm,
        const bf16_t* __restrict__ memT, float* __restrict__ uq) {
    int lane = threadIdx.x & 31, wave = threadIdx.x >> 5;
    int m0 = (blockIdx.y * 8 + wave) * 16;
    int n0 = blockIdx.x * 64;
    int kb = (lane >> 4) * 8;
    const float*  ap = sfxm + (size_t)(m0 + (lane & 15)) * MSLOTS + kb;
    const bf16_t* bp = memT + (size_t)(n0 + (lane & 15)) * MSLOTS + kb;
    v8f acc0 = {}, acc1 = {}, acc2 = {}, acc3 = {};
    for (int ks = 0; ks < MSLOTS / 32; ++ks) {
        int k0 = ks * 32;
        float4 f0 = *(const float4*)(ap + k0);
        float4 f1 = *(const float4*)(ap + k0 + 4);
        float4 f2 = *(const float4*)(ap + k0 + 16);
        float4 f3 = *(const float4*)(ap + k0 + 20);
        ABFrag A, B0, B1, B2, B3;
        A.v[0]  = (bf16_t)f0.x; A.v[1]  = (bf16_t)f0.y; A.v[2]  = (bf16_t)f0.z; A.v[3]  = (bf16_t)f0.w;
        A.v[4]  = (bf16_t)f1.x; A.v[5]  = (bf16_t)f1.y; A.v[6]  = (bf16_t)f1.z; A.v[7]  = (bf16_t)f1.w;
        A.v[8]  = (bf16_t)f2.x; A.v[9]  = (bf16_t)f2.y; A.v[10] = (bf16_t)f2.z; A.v[11] = (bf16_t)f2.w;
        A.v[12] = (bf16_t)f3.x; A.v[13] = (bf16_t)f3.y; A.v[14] = (bf16_t)f3.z; A.v[15] = (bf16_t)f3.w;
        B0.u[0] = *(const uint4*)(bp + k0);
        B0.u[1] = *(const uint4*)(bp + k0 + 16);
        B1.u[0] = *(const uint4*)(bp + 16 * MSLOTS + k0);
        B1.u[1] = *(const uint4*)(bp + 16 * MSLOTS + k0 + 16);
        B2.u[0] = *(const uint4*)(bp + 32 * MSLOTS + k0);
        B2.u[1] = *(const uint4*)(bp + 32 * MSLOTS + k0 + 16);
        B3.u[0] = *(const uint4*)(bp + 48 * MSLOTS + k0);
        B3.u[1] = *(const uint4*)(bp + 48 * MSLOTS + k0 + 16);
        acc0 = __builtin_amdgcn_wmma_f32_16x16x32_bf16(false, A.v, false, B0.v, (short)0, acc0, false, false);
        acc1 = __builtin_amdgcn_wmma_f32_16x16x32_bf16(false, A.v, false, B1.v, (short)0, acc1, false, false);
        acc2 = __builtin_amdgcn_wmma_f32_16x16x32_bf16(false, A.v, false, B2.v, (short)0, acc2, false, false);
        acc3 = __builtin_amdgcn_wmma_f32_16x16x32_bf16(false, A.v, false, B3.v, (short)0, acc3, false, false);
    }
    int ro = m0 + ((lane >> 4) << 3);
    int co = n0 + (lane & 15);
#pragma unroll
    for (int r = 0; r < 8; ++r) {
        size_t rb = (size_t)(ro + r) * (2 * DDIM) + DDIM + co;
        uq[rb]      = acc0[r];
        uq[rb + 16] = acc1[r];
        uq[rb + 32] = acc2[r];
        uq[rb + 48] = acc3[r];
    }
}

// ------------- losses -------------
__global__ void __launch_bounds__(256) k_loss(const float* __restrict__ qf32,
        const float* __restrict__ memory, const int* __restrict__ t1, const int* __restrict__ t2,
        float* __restrict__ loss) {
    int lane = threadIdx.x & 31, wave = threadIdx.x >> 5;
    int j = blockIdx.x * 8 + wave;
    const float* q   = qf32   + (size_t)j * DDIM;
    const float* pos = memory + (size_t)t1[j] * DDIM;
    const float* neg = memory + (size_t)t2[j] * DDIM;
    float g = 0.f, dp = 0.f, dn = 0.f;
    int c = lane * 8;
#pragma unroll
    for (int i = 0; i < 8; ++i) {
        float qv = q[c + i], pv = pos[c + i], nv = neg[c + i];
        float d0 = qv - pv;           g  += d0 * d0;
        float d1 = d0 + 1e-6f;        dp += d1 * d1;
        float d2 = qv - nv + 1e-6f;   dn += d2 * d2;
    }
    g = wave_sum(g); dp = wave_sum(dp); dn = wave_sum(dn);
    if (lane == 0) {
        atomicAdd(&loss[0], g * (1.f / ((float)BS * (float)DDIM)));
        float hinge = fmaxf(sqrtf(dp) - sqrtf(dn) + 1.f, 0.f);
        atomicAdd(&loss[1], hinge * (1.f / (float)BS));
    }
}

extern "C" void kernel_launch(void* const* d_in, const int* in_sizes, int n_in,
                              void* d_out, int out_size, void* d_ws, size_t ws_size,
                              hipStream_t stream) {
    (void)in_sizes; (void)n_in; (void)out_size; (void)ws_size;
    const float* query  = (const float*)d_in[0];   // [BS, D]
    const float* memory = (const float*)d_in[1];   // [M, D]
    float* out = (float*)d_out;

    float* uq   = out;                               // [BS, 2D]
    float* umem = uq   + (size_t)BS * 2 * DDIM;      // [M, D]
    float* sfxq = umem + (size_t)MSLOTS * DDIM;      // [BS, M] (raw score first, softmaxed in place)
    float* sfxm = sfxq + (size_t)BS * MSLOTS;        // [BS, M]
    float* loss = sfxm + (size_t)BS * MSLOTS;        // 2 scalars

    char* ws = (char*)d_ws; size_t o = 0;
    float*  qf32  = (float*) (ws + o); o += (size_t)BS * DDIM * 4;
    bf16_t* qbf   = (bf16_t*)(ws + o); o += (size_t)BS * DDIM * 2;
    bf16_t* membf = (bf16_t*)(ws + o); o += (size_t)MSLOTS * DDIM * 2;
    bf16_t* memT  = (bf16_t*)(ws + o); o += (size_t)MSLOTS * DDIM * 2;
    float*  pmx   = (float*) (ws + o); o += (size_t)16 * MSLOTS * 4;
    float*  ps    = (float*) (ws + o); o += (size_t)16 * MSLOTS * 4;
    float*  cms   = (float*) (ws + o); o += (size_t)MSLOTS * 4;
    float*  csum  = (float*) (ws + o); o += (size_t)MSLOTS * 4;
    float*  rmax  = (float*) (ws + o); o += (size_t)BS * 4;
    float*  rsum  = (float*) (ws + o); o += (size_t)BS * 4;
    int*    t1    = (int*)   (ws + o); o += (size_t)BS * 4;
    int*    t2    = (int*)   (ws + o); o += (size_t)BS * 4;
    float*  wbuf  = (float*) (ws + o); o += (size_t)BS * 4;
    int*    midx  = (int*)   (ws + o); o += (size_t)BS * 4;
    float*  qupd  = (float*) (ws + o); o += (size_t)MSLOTS * DDIM * 4;

    k_init       <<<dim3((MSLOTS * DDIM) / 256), 256, 0, stream>>>(qupd, loss);
    k_norm_query <<<dim3(BS / 8),               256, 0, stream>>>(query, qf32, qbf, uq);
    k_prep_mem   <<<dim3((MSLOTS * DDIM) / 256), 256, 0, stream>>>(memory, membf, memT);
    k_gemm_score <<<dim3(MSLOTS / 64, BS / 128), 256, 0, stream>>>(qbf, membf, sfxq);
    k_col_partial<<<dim3(MSLOTS / 256, 16),     256, 0, stream>>>(sfxq, pmx, ps);
    k_col_merge  <<<dim3(MSLOTS / 256),         256, 0, stream>>>(pmx, ps, cms, csum);
    k_row_stats  <<<dim3(BS / 8),               256, 0, stream>>>(sfxq, rmax, rsum, t1, t2);
    k_softmax    <<<dim3(BS / 8),               256, 0, stream>>>(sfxq, sfxm, cms, csum, rmax, rsum, wbuf, midx);
    k_segsum     <<<dim3(BS),                   256, 0, stream>>>(qf32, wbuf, midx, qupd);
    k_update_mem <<<dim3(MSLOTS / 8),           256, 0, stream>>>(qupd, memory, umem);
    k_gemm_read  <<<dim3(DDIM / 64, BS / 128),  256, 0, stream>>>(sfxm, memT, uq);
    k_loss       <<<dim3(BS / 8),               256, 0, stream>>>(qf32, memory, t1, t2, loss);
}
